// LGA_23407571763401
// MI455X (gfx1250) — compile-verified, matrix operand order checked by reading, not run on hardware
//
#include <hip/hip_runtime.h>
#include <hip/hip_bf16.h>

// ---------------------------------------------------------------------------
// Problem constants (from reference): B=4, G=2048, K=24, C=96, R=64, 3 sigmas
//   in_ch = 576, out_dim = 192.  Output [B, 192, G, K] fp32.
// ---------------------------------------------------------------------------
#define NB    4
#define NG    2048
#define NK    24
#define NC    96
#define NR    64
#define INCH  576          // 3 sigmas * 3 dims * 64 refs
#define OUTD  192
#define KT_N  18           // 576 / 32  (bf16 WMMA K=32 slices)
#define NT_N  12           // 192 / 16
#define MT_N  3            // 48 rows / 16  (2 groups * 24 neighbors)
#define GROUPS_PER_WG 2
#define THREADS 192        // 6 waves (wave32)
#define EPSF  1e-5f

typedef __bf16  v16bf __attribute__((ext_vector_type(16)));
typedef float   v8f   __attribute__((ext_vector_type(8)));

// K-index inside a 32-wide slice for (lane-half, element e) of a 16-bit
// WMMA fragment (per CDNA5 ISA 16-bit A-matrix 16x32 table; B uses the
// transposed analog with N striped across lanes).
__device__ __forceinline__ int frag_kk(int half, int e) {
    int v = e >> 1, hw = e & 1;
    return (half << 3) + ((v & 4) << 2) + ((v & 3) << 1) + hw;
}

__device__ __forceinline__ unsigned short bf16_bits(float f) {
    return __builtin_bit_cast(unsigned short, (__bf16)f);
}

// ---------------------------------------------------------------------------
// Prep: fuse_w [192 x 576] f32  ->  bf16 WMMA B fragments in d_ws.
// Tile (kt, nt) is 32(K) x 16(N); storage: ws[((kt*12+nt)*32 + lane)*16 + e]
// ---------------------------------------------------------------------------
__global__ __launch_bounds__(256) void lga_prep_w(const float* __restrict__ w,
                                                  unsigned short* __restrict__ ws) {
    int idx = blockIdx.x * 256 + threadIdx.x;          // 110592 total
    if (idx >= KT_N * NT_N * 512) return;
    int e    = idx & 15;
    int lane = (idx >> 4) & 31;
    int tile = idx >> 9;
    int nt   = tile % NT_N;
    int kt   = tile / NT_N;
    int o    = nt * 16 + (lane & 15);
    int i    = kt * 32 + frag_kk(lane >> 4, e);
    ws[idx]  = bf16_bits(w[o * INCH + i]);
}

// ---------------------------------------------------------------------------
// Main fused kernel: one workgroup = 2 (b,g) groups (48 rows).
// Wave w owns N-tile pair {2w, 2w+1} x all 3 M-tiles -> each B fragment is
// fetched exactly once per workgroup (L2-resident weights, 1x traffic).
// ---------------------------------------------------------------------------
__global__ __launch_bounds__(THREADS) void lga_fused(
    const float* __restrict__ lc_x,
    const float* __restrict__ knn_xyz,
    const float* __restrict__ knn_x,
    const unsigned short* __restrict__ wsB,
    const float* __restrict__ fuse_b,
    const float* __restrict__ ln_g,
    const float* __restrict__ ln_b,
    float* __restrict__ out) {

    __shared__ alignas(32) unsigned int lgaA[KT_N * MT_N * 256]; // A frags (bf16x2 words)
    __shared__ float lgaY[48 * OUTD];                            // x + pos tile
    __shared__ float lgaXYZ[48 * 3];                             // normalized coords
    __shared__ float lgaMu[48];
    __shared__ float lgaRs[48];

    const int t    = threadIdx.x;
    const int lane = t & 31;
    const int w    = t >> 5;                // wave id 0..5
    const int bg0  = blockIdx.x * GROUPS_PER_WG;

    // ---- Phase 1a: per-(group, channel) K-normalization of knn_x into y tile,
    //      plus broadcast of lc_x into columns 96..191.
    {
        int gl = t / NC, c = t % NC;        // 192 threads cover 2*96 columns
        const float* src = knn_x + ((size_t)(bg0 + gl) * NK) * NC + c;
        float tmp[NK];
        float s1 = 0.f;
        #pragma unroll
        for (int k = 0; k < NK; ++k) { tmp[k] = src[k * NC]; s1 += tmp[k]; }
        float mu = s1 * (1.f / NK);
        float s2 = 0.f;
        #pragma unroll
        for (int k = 0; k < NK; ++k) { float d = tmp[k] - mu; s2 += d * d; }
        float inv = 1.f / (sqrtf(fmaxf(s2 * (1.f / (NK - 1)), 0.f)) + EPSF);
        float lc  = lc_x[(size_t)(bg0 + gl) * NC + c];
        #pragma unroll
        for (int k = 0; k < NK; ++k) {
            int row = gl * NK + k;
            lgaY[row * OUTD + c]      = (tmp[k] - mu) * inv;
            lgaY[row * OUTD + NC + c] = lc;
        }
    }
    // ---- Phase 1b: K-normalization of knn_xyz (6 columns).
    if (t < GROUPS_PER_WG * 3) {
        int gl = t / 3, d = t % 3;
        const float* src = knn_xyz + ((size_t)(bg0 + gl) * NK) * 3 + d;
        float tmp[NK];
        float s1 = 0.f;
        #pragma unroll
        for (int k = 0; k < NK; ++k) { tmp[k] = src[k * 3]; s1 += tmp[k]; }
        float mu = s1 * (1.f / NK);
        float s2 = 0.f;
        #pragma unroll
        for (int k = 0; k < NK; ++k) { float dd = tmp[k] - mu; s2 += dd * dd; }
        float inv = 1.f / (sqrtf(fmaxf(s2 * (1.f / (NK - 1)), 0.f)) + EPSF);
        #pragma unroll
        for (int k = 0; k < NK; ++k)
            lgaXYZ[(gl * NK + k) * 3 + d] = (tmp[k] - mu) * inv;
    }
    __syncthreads();

    // ---- Phase 2: build RBF-encoded A fragments (bf16) in LDS.
    //      13824 packed-pair words; 72 per thread; conflict-free b32 stores.
    #pragma unroll 4
    for (int it = 0; it < 72; ++it) {
        int idx   = it * THREADS + t;
        int epair = idx & 7;
        int l     = (idx >> 3) & 31;
        int tile  = idx >> 8;               // 0..53  == kt*3 + mt
        int mt    = tile % MT_N;
        int kt    = tile / MT_N;
        int m     = mt * 16 + (l & 15);     // row 0..47
        int i0    = kt * 32 + frag_kk(l >> 4, epair * 2);   // even -> r0 even
        int s     = (i0 >= 384) ? 2 : (i0 >= 192) ? 1 : 0;
        int rem   = i0 - s * 192;
        int d     = rem >> 6;
        int r0    = rem & 63;
        float coef = (s == 0) ? 3.125f : (s == 1) ? 1.3888888888888888f : 0.78125f;
        float x    = lgaXYZ[m * 3 + d];
        float ref0 = -1.f + (2.f / 63.f) * (float)r0;
        float t0   = x - ref0;
        float t1   = x - (ref0 + 2.f / 63.f);
        unsigned int p = (unsigned int)bf16_bits(__expf(-t0 * t0 * coef)) |
                         ((unsigned int)bf16_bits(__expf(-t1 * t1 * coef)) << 16);
        lgaA[tile * 256 + l * 8 + epair] = p;
    }
    __syncthreads();

    // ---- Phase 3: bf16 WMMA GEMM  pos = enc(48x576) * W^T(576x192)
    //      wave w: nt in {2w, 2w+1}, mt in {0,1,2}  -> 6 accumulators.
    const int nt0 = w * 2;
    v8f acc[6] = {};                         // acc[mt*2 + jn]
    for (int kt = 0; kt < KT_N; ++kt) {
        const v16bf* bp = reinterpret_cast<const v16bf*>(wsB) +
                          ((size_t)(kt * NT_N + nt0) * 32 + lane);
        if (kt + 1 < KT_N)
            __builtin_prefetch(reinterpret_cast<const v16bf*>(wsB) +
                               ((size_t)((kt + 1) * NT_N + nt0) * 32 + lane), 0, 3);
        v16bf b0 = bp[0];
        v16bf b1 = bp[32];
        #pragma unroll
        for (int mt = 0; mt < MT_N; ++mt) {
            v16bf a = *reinterpret_cast<const v16bf*>(
                &lgaA[(kt * MT_N + mt) * 256 + lane * 8]);
            acc[mt * 2 + 0] = __builtin_amdgcn_wmma_f32_16x16x32_bf16(
                false, a, false, b0, (short)0, acc[mt * 2 + 0], false, false);
            acc[mt * 2 + 1] = __builtin_amdgcn_wmma_f32_16x16x32_bf16(
                false, a, false, b1, (short)0, acc[mt * 2 + 1], false, false);
        }
    }

    // ---- Phase 4: y += pos + bias  (disjoint regions per wave, no races)
    {
        int half = lane >> 4, nlo = lane & 15;
        #pragma unroll
        for (int jn = 0; jn < 2; ++jn) {
            int col  = (nt0 + jn) * 16 + nlo;
            float bb = fuse_b[col];
            #pragma unroll
            for (int mt = 0; mt < MT_N; ++mt) {
                #pragma unroll
                for (int r = 0; r < 8; ++r) {
                    int row = mt * 16 + half * 8 + r;
                    lgaY[row * OUTD + col] += acc[mt * 2 + jn][r] + bb;
                }
            }
        }
    }
    __syncthreads();

    // ---- Phase 5: LayerNorm stats, 4 lanes per row, shfl reduction.
    {
        int row = t >> 2, q = t & 3;
        float s1 = 0.f, s2 = 0.f;
        #pragma unroll
        for (int c = 0; c < 48; ++c) {
            float v = lgaY[row * OUTD + q * 48 + c];
            s1 += v; s2 += v * v;
        }
        s1 += __shfl_xor(s1, 1, 32); s2 += __shfl_xor(s2, 1, 32);
        s1 += __shfl_xor(s1, 2, 32); s2 += __shfl_xor(s2, 2, 32);
        float mu  = s1 * (1.f / OUTD);
        float var = fmaxf(s2 * (1.f / OUTD) - mu * mu, 0.f);
        if (q == 0) { lgaMu[row] = mu; lgaRs[row] = rsqrtf(var + EPSF); }
    }
    __syncthreads();

    // ---- Phase 6: normalize + transpose-write. One (group, o) run = 24
    //      contiguous floats = 6 x global_store_b128 per lane.
    #pragma unroll
    for (int rep = 0; rep < 2; ++rep) {
        int run = t + rep * THREADS;        // 0..383
        int gl  = (run >= OUTD) ? 1 : 0;
        int o   = run - gl * OUTD;
        int bg  = bg0 + gl;
        int b   = bg >> 11;                 // / 2048
        int g   = bg & 2047;
        float ga = ln_g[o], be = ln_b[o];
        float4* op = reinterpret_cast<float4*>(
            out + (((size_t)b * OUTD + o) * NG + g) * NK);
        #pragma unroll
        for (int kb = 0; kb < 6; ++kb) {
            float4 vv;
            #pragma unroll
            for (int c = 0; c < 4; ++c) {
                int row = gl * NK + kb * 4 + c;
                (&vv.x)[c] = (lgaY[row * OUTD + o] - lgaMu[row]) * lgaRs[row] * ga + be;
            }
            op[kb] = vv;
        }
    }
}

// ---------------------------------------------------------------------------
extern "C" void kernel_launch(void* const* d_in, const int* in_sizes, int n_in,
                              void* d_out, int out_size, void* d_ws, size_t ws_size,
                              hipStream_t stream) {
    // setup_inputs order: lc_xyz, lc_x, knn_xyz, knn_x, fuse_w, fuse_b, ln_gamma, ln_beta
    const float* lc_x    = (const float*)d_in[1];
    const float* knn_xyz = (const float*)d_in[2];
    const float* knn_x   = (const float*)d_in[3];
    const float* fuse_w  = (const float*)d_in[4];
    const float* fuse_b  = (const float*)d_in[5];
    const float* ln_g    = (const float*)d_in[6];
    const float* ln_b    = (const float*)d_in[7];
    float* out           = (float*)d_out;
    unsigned short* wsB  = (unsigned short*)d_ws;   // 221184 bytes of bf16 W fragments

    // Re-pack weights into WMMA B-fragment order (deterministic, every call).
    lga_prep_w<<<(KT_N * NT_N * 512 + 255) / 256, 256, 0, stream>>>(fuse_w, wsB);

    // One workgroup per 2 (b,g) groups.
    const int nblocks = (NB * NG) / GROUPS_PER_WG;  // 4096
    lga_fused<<<nblocks, THREADS, 0, stream>>>(lc_x, knn_xyz, knn_x, wsB,
                                               fuse_b, ln_g, ln_b, out);
}
